// Aligner_78950088835164
// MI455X (gfx1250) — matrix operand order repaired and checked by reading, not compiled, not run on hardware
//
#include <hip/hip_runtime.h>
#include <hip/hip_bf16.h>

typedef __attribute__((ext_vector_type(16))) _Float16 v16h;
typedef __attribute__((ext_vector_type(8)))  _Float16 v8h;
typedef __attribute__((ext_vector_type(8)))  float    v8f;

#define NEGINF (-__builtin_inff())

// b=64, t_x=256, t_y=1024, dim_in=80, dim_hidden=512, attn_c=80
static constexpr int BB  = 64;
static constexpr int TX  = 256;
static constexpr int TY  = 1024;
static constexpr int NX  = BB * TX;     // 16384 key tokens
static constexpr int NY  = BB * TY;     // 65536 query tokens

// ---------------------------------------------------------------------------
// Packing kernels
// ---------------------------------------------------------------------------

// conv weight (Cout, Cin, K) f32 -> [Cout][Kp] f16 with k = cin*K + tap, zero pad.
__global__ void pack_w(const float* __restrict__ w, _Float16* __restrict__ out,
                       int Cout, int Cin, int K, int Kp) {
  int gid = blockIdx.x * 256 + threadIdx.x;
  if (gid >= Cout * Kp) return;
  int m = gid / Kp, kk = gid % Kp;
  float v = 0.f;
  if (kk < Cin * K) v = w[(m * Cin + kk / K) * K + kk % K];
  out[gid] = (_Float16)v;
}

// x (64,512,256) f32 -> imX[(b*256+t)][cin*3+tap] f16 (Kp=1536); b128 stores
__global__ void im2col_x(const float* __restrict__ x, _Float16* __restrict__ out) {
  long gid = (long)blockIdx.x * 256 + threadIdx.x;   // NX*192 8-half chunks
  int n = (int)(gid / 192), c8 = (int)(gid % 192);
  int b = n >> 8, t = n & 255;
  v8h hv;
#pragma unroll
  for (int r = 0; r < 8; ++r) {
    int kk = c8 * 8 + r;
    int cin = kk / 3, tap = kk % 3;
    int tt = t + tap - 1;
    float v = (tt >= 0 && tt < TX) ? x[((size_t)(b * 512 + cin)) * TX + tt] : 0.f;
    hv[r] = (_Float16)v;
  }
  *(v8h*)(out + gid * 8) = hv;
}

// y (64,1024,80) f32 -> imY[(b*1024+t)][cin*3+tap] f16, Kp=256 (real 240)
__global__ void im2col_y(const float* __restrict__ y, _Float16* __restrict__ out) {
  long gid = (long)blockIdx.x * 256 + threadIdx.x;   // NY*32 8-half chunks
  int n = (int)(gid / 32), c8 = (int)(gid % 32);
  int b = n >> 10, t = n & 1023;
  v8h hv;
#pragma unroll
  for (int r = 0; r < 8; ++r) {
    int kk = c8 * 8 + r;
    float v = 0.f;
    if (kk < 240) {
      int cin = kk / 3, tap = kk % 3;
      int tt = t + tap - 1;
      if (tt >= 0 && tt < TY) v = y[((size_t)(b * TY + tt)) * 80 + cin];
    }
    hv[r] = (_Float16)v;
  }
  *(v8h*)(out + gid * 8) = hv;
}

__global__ void zero_f32v4(float4* __restrict__ p, long n4) {
  long i = (long)blockIdx.x * blockDim.x + threadIdx.x;
  if (i < n4) p[i] = make_float4(0.f, 0.f, 0.f, 0.f);
}

// sum of squares over ld f16 cols (pads are zero); v8h loads
__global__ void sqsum(const _Float16* __restrict__ a, float* __restrict__ out,
                      int n, int ld) {
  int i = blockIdx.x * 256 + threadIdx.x;
  if (i >= n) return;
  const v8h* p = (const v8h*)(a + (size_t)i * ld);
  float s = 0.f;
  for (int c = 0; c < ld / 8; ++c) {
    v8h h = p[c];
#pragma unroll
    for (int r = 0; r < 8; ++r) { float v = (float)h[r]; s += v * v; }
  }
  out[i] = s;
}

// ---------------------------------------------------------------------------
// Tiled WMMA GEMM: C[N][ldo](f16) = act(A[M][Kp] * B[N][Kp]^T + bias)
// Block = 256 threads = 8 waves (4 M x 2 N), tile 128x64, K-step 32.
// Each wave: 2x2 tiles of v_wmma_f32_16x16x32_f16.
// Rows of out in [M, ldo) get exact zeros (A rows >= M zero-filled in LDS,
// bias guarded), giving zero K-padding for the next GEMM.
// ---------------------------------------------------------------------------
template <bool RELU>
__global__ void __launch_bounds__(256)
gemm_bias_act(const _Float16* __restrict__ A, const _Float16* __restrict__ B,
              const float* __restrict__ bias, _Float16* __restrict__ outH,
              int M, int N, int Kp, int ldo) {
  __shared__ _Float16 As[128][40];
  __shared__ _Float16 Bs[64][40];
  int tid  = threadIdx.x;
  int lane = tid & 31, w = tid >> 5;
  int wm = w & 3, wn = w >> 2;
  int lm = lane & 15, lh = lane >> 4;
  int M0 = blockIdx.y * 128, N0 = blockIdx.x * 64;

  v8f acc[2][2] = {};

  for (int k0 = 0; k0 < Kp; k0 += 32) {
#pragma unroll
    for (int it = 0; it < 2; ++it) {
      int idx = tid + it * 256;
      int r = idx >> 2, c = idx & 3;
      int gm = M0 + r;
      v8h val = {};
      if (gm < M) val = *(const v8h*)(A + (size_t)gm * Kp + k0 + c * 8);
      *(v8h*)&As[r][c * 8] = val;
    }
    {
      int r = tid >> 2, c = tid & 3;
      int gn = N0 + r;
      v8h val = {};
      if (gn < N) val = *(const v8h*)(B + (size_t)gn * Kp + k0 + c * 8);
      *(v8h*)&Bs[r][c * 8] = val;
    }
    __syncthreads();

    v16h af[2], bf[2];
#pragma unroll
    for (int mi = 0; mi < 2; ++mi) {           // A frag: row m, chunks k=lh*8, 16+lh*8
      const _Float16* p = &As[wm * 32 + mi * 16 + lm][lh * 8];
      union { v16h v; v8h h[2]; } u;
      u.h[0] = *(const v8h*)p;
      u.h[1] = *(const v8h*)(p + 16);
      af[mi] = u.v;
    }
#pragma unroll
    for (int ni = 0; ni < 2; ++ni) {           // B frag: col n, chunks k=lh*8, 16+lh*8
      const _Float16* p = &Bs[wn * 32 + ni * 16 + lm][lh * 8];
      union { v16h v; v8h h[2]; } u;
      u.h[0] = *(const v8h*)p;
      u.h[1] = *(const v8h*)(p + 16);
      bf[ni] = u.v;
    }
#pragma unroll
    for (int mi = 0; mi < 2; ++mi)
#pragma unroll
      for (int ni = 0; ni < 2; ++ni)
        acc[mi][ni] = __builtin_amdgcn_wmma_f32_16x16x32_f16(
            false, af[mi], false, bf[ni], (short)0, acc[mi][ni], false, false);
    __syncthreads();
  }

  // Epilogue: 8 contiguous m-halfs per lane -> one 16B store per tile.
#pragma unroll
  for (int mi = 0; mi < 2; ++mi)
#pragma unroll
    for (int ni = 0; ni < 2; ++ni) {
      int gm0 = M0 + wm * 32 + mi * 16 + 8 * lh;   // multiple of 8
      int gn  = N0 + wn * 32 + ni * 16 + lm;
      if (gm0 < ldo && gn < N) {                   // ldo multiple of 16
        v8h hv;
#pragma unroll
        for (int r = 0; r < 8; ++r) {
          int gm = gm0 + r;
          float v = acc[mi][ni][r] + (gm < M ? bias[gm] : 0.f);
          if (RELU) v = fmaxf(v, 0.f);
          hv[r] = (_Float16)v;
        }
        *(v8h*)(outH + (size_t)gn * ldo + gm0) = hv;
      }
    }
}

// ---------------------------------------------------------------------------
// Distance GEMM (per batch): dot = q . k over Kp=96, then
// logp[ty][tx] = mask[tx] ? sqrt(max(|q|^2+|k|^2-2 dot, 0)) : -inf
// M=1024, N=256, grid (4, 8, 64). f32 stores coalesced across lanes.
// ---------------------------------------------------------------------------
__global__ void __launch_bounds__(256)
gemm_dist(const _Float16* __restrict__ Qm, const _Float16* __restrict__ Km,
          const float* __restrict__ qsq, const float* __restrict__ ksq,
          const float* __restrict__ xmask, float* __restrict__ outF) {
  constexpr int Kp = 96;
  int b = blockIdx.z;
  const _Float16* A = Qm + (size_t)b * TY * Kp;
  const _Float16* B = Km + (size_t)b * TX * Kp;
  const float* qs = qsq + b * TY;
  const float* ks = ksq + b * TX;
  const float* xm = xmask + b * TX;
  float* out = outF + (size_t)b * TY * TX;

  __shared__ _Float16 As[128][40];
  __shared__ _Float16 Bs[64][40];
  int tid  = threadIdx.x;
  int lane = tid & 31, w = tid >> 5;
  int wm = w & 3, wn = w >> 2;
  int lm = lane & 15, lh = lane >> 4;
  int M0 = blockIdx.y * 128, N0 = blockIdx.x * 64;

  v8f acc[2][2] = {};

  for (int k0 = 0; k0 < Kp; k0 += 32) {
#pragma unroll
    for (int it = 0; it < 2; ++it) {
      int idx = tid + it * 256;
      int r = idx >> 2, c = idx & 3;
      *(v8h*)&As[r][c * 8] = *(const v8h*)(A + (size_t)(M0 + r) * Kp + k0 + c * 8);
    }
    {
      int r = tid >> 2, c = tid & 3;
      *(v8h*)&Bs[r][c * 8] = *(const v8h*)(B + (size_t)(N0 + r) * Kp + k0 + c * 8);
    }
    __syncthreads();

    v16h af[2], bf[2];
#pragma unroll
    for (int mi = 0; mi < 2; ++mi) {
      const _Float16* p = &As[wm * 32 + mi * 16 + lm][lh * 8];
      union { v16h v; v8h h[2]; } u;
      u.h[0] = *(const v8h*)p; u.h[1] = *(const v8h*)(p + 16);
      af[mi] = u.v;
    }
#pragma unroll
    for (int ni = 0; ni < 2; ++ni) {
      const _Float16* p = &Bs[wn * 32 + ni * 16 + lm][lh * 8];
      union { v16h v; v8h h[2]; } u;
      u.h[0] = *(const v8h*)p; u.h[1] = *(const v8h*)(p + 16);
      bf[ni] = u.v;
    }
#pragma unroll
    for (int mi = 0; mi < 2; ++mi)
#pragma unroll
      for (int ni = 0; ni < 2; ++ni)
        acc[mi][ni] = __builtin_amdgcn_wmma_f32_16x16x32_f16(
            false, af[mi], false, bf[ni], (short)0, acc[mi][ni], false, false);
    __syncthreads();
  }

#pragma unroll
  for (int mi = 0; mi < 2; ++mi)
#pragma unroll
    for (int ni = 0; ni < 2; ++ni)
#pragma unroll
      for (int r = 0; r < 8; ++r) {
        int gm = M0 + wm * 32 + mi * 16 + r + 8 * lh;
        int gn = N0 + wn * 32 + ni * 16 + lm;
        float d2 = qs[gm] + ks[gn] - 2.f * acc[mi][ni][r];
        float s = sqrtf(fmaxf(d2, 0.f));
        out[(size_t)gm * TX + gn] = (xm[gn] != 0.f) ? s : NEGINF;
      }
}

// ---------------------------------------------------------------------------
// Softmax over t_x (256) per (b, t_y) row; transposed store into soft(b,t_x,t_y)
// ---------------------------------------------------------------------------
__global__ void softmax_rows(const float* __restrict__ logp, float* __restrict__ soft) {
  int row = blockIdx.x;            // b*1024 + ty
  int b = row >> 10, ty = row & 1023;
  int i = threadIdx.x;             // tx
  float v = logp[(size_t)row * TX + i];
  __shared__ float red[256];
  red[i] = v; __syncthreads();
  for (int s = 128; s > 0; s >>= 1) { if (i < s) red[i] = fmaxf(red[i], red[i + s]); __syncthreads(); }
  float mx = red[0]; __syncthreads();
  float e = __expf(v - mx);
  red[i] = e; __syncthreads();
  for (int s = 128; s > 0; s >>= 1) { if (i < s) red[i] += red[i + s]; __syncthreads(); }
  float sum = red[0];
  soft[((size_t)(b * TX + i)) * TY + ty] = e / sum;
}

// ---------------------------------------------------------------------------
// Monotonic alignment Viterbi forward: one block per batch, v in LDS.
// dir[(b*1024 + j)*256 + i] = (v[i] >= v[i-1])
// ---------------------------------------------------------------------------
__global__ void dp_forward(const float* __restrict__ soft,
                           const float* __restrict__ xmask,
                           const float* __restrict__ ymask,
                           unsigned char* __restrict__ dir) {
  int b = blockIdx.x, i = threadIdx.x;
  __shared__ float v[256];
  v[i] = 0.f;
  float xm = xmask[b * TX + i];
  __syncthreads();
  for (int j = 0; j < TY; ++j) {
    float vi = v[i];
    float vim1 = (i > 0) ? v[i - 1] : NEGINF;
    __syncthreads();
    int d = (vi >= vim1) ? 1 : 0;
    float vmax = d ? vi : vim1;
    float val = soft[((size_t)(b * TX + i)) * TY + j] * xm * ymask[b * TY + j];
    v[i] = (i <= j) ? (vmax + val) : NEGINF;
    dir[((size_t)(b * TY + j)) * TX + i] = (unsigned char)d;
    __syncthreads();
  }
}

// Backtrace: one block per batch; thread 0 walks, all threads emit hard counts.
__global__ void backtrace(const unsigned char* __restrict__ dir,
                          const float* __restrict__ xmask,
                          float* __restrict__ path, int* __restrict__ hard) {
  int b = blockIdx.x, i = threadIdx.x;
  __shared__ int cnt[256];
  cnt[i] = 0; __syncthreads();
  if (i == 0) {
    float s = 0.f;
    for (int t = 0; t < TX; ++t) s += xmask[b * TX + t];
    int idx = (int)s - 1; if (idx < 0) idx = 0; if (idx > TX - 1) idx = TX - 1;
    for (int j = TY - 1; j >= 0; --j) {
      path[((size_t)(b * TX + idx)) * TY + j] = 1.0f;
      cnt[idx]++;
      int d = dir[((size_t)(b * TY + j)) * TX + idx];
      idx += d - 1;
      if (idx < 0) idx = 0;
    }
  }
  __syncthreads();
  hard[b * TX + i] = cnt[i];
}

// ---------------------------------------------------------------------------
// Workspace layout (bytes); all offsets 16B aligned.
// ---------------------------------------------------------------------------
static constexpr size_t OW1   = 0;                                   // 1024x1536 f16
static constexpr size_t OW2   = OW1 + (size_t)1024 * 1536 * 2;       // 80x1024
static constexpr size_t OW3   = OW2 + (size_t)80 * 1024 * 2;         // 160x256
static constexpr size_t OW4   = OW3 + (size_t)160 * 256 * 2;         // 80x160
static constexpr size_t OW5   = OW4 + (size_t)80 * 160 * 2;          // 80x96
static constexpr size_t OIMX  = OW5 + (size_t)80 * 96 * 2;           // 16384x1536 f16
static constexpr size_t OACT1 = OIMX + (size_t)NX * 1536 * 2;        // 16384x1024 f16 (reused: 65536x160 f16)
static constexpr size_t OKM   = OACT1 + (size_t)NX * 1024 * 2;       // 16384x96 f16
static constexpr size_t OKSQ  = OKM + (size_t)NX * 96 * 2;           // 16384 f32
static constexpr size_t OQSQ  = OKSQ + (size_t)NX * 4;               // 65536 f32
static constexpr size_t OIMY  = OQSQ + (size_t)NY * 4;               // 65536x256 f16 (reused: dir bytes)
static constexpr size_t OQ2   = OIMY + (size_t)NY * 256 * 2;         // 65536x96 f16
static constexpr size_t OQM   = OQ2 + (size_t)NY * 96 * 2;           // 65536x96 f16

extern "C" void kernel_launch(void* const* d_in, const int* in_sizes, int n_in,
                              void* d_out, int out_size, void* d_ws, size_t ws_size,
                              hipStream_t stream) {
  const float* x      = (const float*)d_in[0];
  const float* x_mask = (const float*)d_in[1];
  const float* y      = (const float*)d_in[2];
  const float* y_mask = (const float*)d_in[3];
  const float* kw1 = (const float*)d_in[4];
  const float* kb1 = (const float*)d_in[5];
  const float* kw2 = (const float*)d_in[6];
  const float* kb2 = (const float*)d_in[7];
  const float* qw1 = (const float*)d_in[8];
  const float* qb1 = (const float*)d_in[9];
  const float* qw2 = (const float*)d_in[10];
  const float* qb2 = (const float*)d_in[11];
  const float* qw3 = (const float*)d_in[12];
  const float* qb3 = (const float*)d_in[13];

  // outputs, concatenated flat: hard(i32), soft(f32), logp(f32), mas(f32)
  int*   hard = (int*)d_out;
  float* soft = (float*)d_out + (size_t)BB * TX;
  float* logp = soft + (size_t)BB * TX * TY;
  float* mas  = logp + (size_t)BB * TY * TX;

  char* ws = (char*)d_ws;
  _Float16* w1   = (_Float16*)(ws + OW1);
  _Float16* w2   = (_Float16*)(ws + OW2);
  _Float16* w3   = (_Float16*)(ws + OW3);
  _Float16* w4   = (_Float16*)(ws + OW4);
  _Float16* w5   = (_Float16*)(ws + OW5);
  _Float16* imX  = (_Float16*)(ws + OIMX);
  _Float16* act1 = (_Float16*)(ws + OACT1);   // keys act, later query act1
  _Float16* kmat = (_Float16*)(ws + OKM);
  float*    ksq  = (float*)(ws + OKSQ);
  float*    qsq  = (float*)(ws + OQSQ);
  _Float16* imY  = (_Float16*)(ws + OIMY);
  unsigned char* dir = (unsigned char*)(ws + OIMY);  // reused after G3
  _Float16* act2 = (_Float16*)(ws + OQ2);
  _Float16* qmat = (_Float16*)(ws + OQM);

  // ---- pack weights ----
  pack_w<<<(1024 * 1536 + 255) / 256, 256, 0, stream>>>(kw1, w1, 1024, 512, 3, 1536);
  pack_w<<<(80 * 1024 + 255) / 256, 256, 0, stream>>>(kw2, w2, 80, 1024, 1, 1024);
  pack_w<<<(160 * 256 + 255) / 256, 256, 0, stream>>>(qw1, w3, 160, 80, 3, 256);
  pack_w<<<(80 * 160 + 255) / 256, 256, 0, stream>>>(qw2, w4, 80, 160, 1, 160);
  pack_w<<<(80 * 96 + 255) / 256, 256, 0, stream>>>(qw3, w5, 80, 80, 1, 96);

  // ---- im2col (8-half chunks) ----
  im2col_x<<<(int)(((long)NX * 192) / 256), 256, 0, stream>>>(x, imX);
  im2col_y<<<(int)(((long)NY * 32) / 256), 256, 0, stream>>>(y, imY);

  // ---- zero alignment_mas ----
  zero_f32v4<<<(int)(((long)BB * TX * TY / 4 + 255) / 256), 256, 0, stream>>>(
      (float4*)mas, (long)BB * TX * TY / 4);

  // ---- keys path ----
  gemm_bias_act<true ><<<dim3(NX / 64, 8), 256, 0, stream>>>(w1, imX,  kb1, act1, 1024, NX, 1536, 1024);
  gemm_bias_act<false><<<dim3(NX / 64, 1), 256, 0, stream>>>(w2, act1, kb2, kmat,   80, NX, 1024,   96);

  // ---- query path (act1 region reused) ----
  gemm_bias_act<true ><<<dim3(NY / 64, 2), 256, 0, stream>>>(w3, imY,  qb1, act1,  160, NY,  256,  160);
  gemm_bias_act<true ><<<dim3(NY / 64, 1), 256, 0, stream>>>(w4, act1, qb2, act2,   80, NY,  160,   96);
  gemm_bias_act<false><<<dim3(NY / 64, 1), 256, 0, stream>>>(w5, act2, qb3, qmat,   80, NY,   96,   96);

  // ---- squared norms ----
  sqsum<<<NX / 256, 256, 0, stream>>>(kmat, ksq, NX, 96);
  sqsum<<<NY / 256, 256, 0, stream>>>(qmat, qsq, NY, 96);

  // ---- distance + logp ----
  gemm_dist<<<dim3(TX / 64, TY / 128, BB), 256, 0, stream>>>(qmat, kmat, qsq, ksq, x_mask, logp);

  // ---- softmax -> alignment_soft (transposed) ----
  softmax_rows<<<BB * TY, 256, 0, stream>>>(logp, soft);

  // ---- monotonic alignment DP + backtrace ----
  dp_forward<<<BB, 256, 0, stream>>>(soft, x_mask, y_mask, dir);
  backtrace<<<BB, 256, 0, stream>>>(dir, x_mask, mas, hard);

  (void)in_sizes; (void)n_in; (void)out_size; (void)ws_size;
}
